// Attention_14448269984093
// MI455X (gfx1250) — compile-verified, matrix operand order checked by reading, not run on hardware
//
#include <hip/hip_runtime.h>
#include <math.h>
#include <stdint.h>

// ---------------- types ----------------
typedef __attribute__((ext_vector_type(16))) _Float16 v16h;
typedef __attribute__((ext_vector_type(4)))  _Float16 v4h;
typedef __attribute__((ext_vector_type(8)))  float    v8f;
typedef __attribute__((ext_vector_type(4)))  float    f32x4;
typedef __attribute__((ext_vector_type(4)))  unsigned int u32x4;
typedef __attribute__((ext_vector_type(8)))  int      i32x8;
typedef __attribute__((ext_vector_type(4)))  int      i32x4;

#define DIM       2048
#define SEQ       4096
#define NHEADS    16
#define HEADDIM   128

static __device__ __forceinline__ v8f v8f_zero() {
  v8f z = {0.f,0.f,0.f,0.f,0.f,0.f,0.f,0.f};
  return z;
}

// Load a 16x32 f16 fragment row (two contiguous 16B chunks) into v16h.
static __device__ __forceinline__ v16h ld_frag(const _Float16* p0, const _Float16* p1) {
  union { v16h v; f32x4 f[2]; } u;
  u.f[0] = *(const f32x4*)p0;
  u.f[1] = *(const f32x4*)p1;
  return u.v;
}

// ---------------- Tensor Data Mover: 2-D tile load Global -> LDS ----------------
// D# per CDNA5 ISA ch.8: group0 = {count, lds_addr, global_addr, type=2},
// group1 = {data_size=1(2B), pad_enable, pad_interval, pad_amount,
//           tensor_dim0/1, tile_dim0/1, tensor_dim0_stride}. Groups 2/3 = 0 (2-D).
// pad_interval code c -> 2^(c+1) dwords between pads; pad_amount code a -> (a+1) dwords.
static __device__ __forceinline__ void tdm_load_2d(
    unsigned lds_off, const void* gaddr,
    unsigned tile_d0, unsigned tile_d1,
    unsigned tensor_d0, unsigned tensor_d1, unsigned stride_elems,
    unsigned pad_interval_code, unsigned pad_amount_code) {
  unsigned long long ga = (unsigned long long)(uintptr_t)gaddr;
  u32x4 g0;
  g0.x = 1u;                                     // count=1, user mode
  g0.y = lds_off;                                // LDS byte address
  g0.z = (unsigned)(ga & 0xffffffffu);           // global_addr[31:0]
  g0.w = (unsigned)((ga >> 32) & 0x01ffffffu) | (2u << 30);  // addr[56:32] | type=2
  i32x8 g1;
  g1[0] = (int)((1u << 16) |                     // data_size = 2 bytes
                (1u << 20) |                     // pad_enable
                (pad_interval_code << 22) |
                (pad_amount_code << 25));
  g1[1] = (int)((tensor_d0 & 0xffffu) << 16);                               // dim0[15:0]
  g1[2] = (int)(((tensor_d0 >> 16) & 0xffffu) | ((tensor_d1 & 0xffffu) << 16));
  g1[3] = (int)(((tensor_d1 >> 16) & 0xffffu) | ((tile_d0 & 0xffffu) << 16));
  g1[4] = (int)(tile_d1 & 0xffffu);              // tile_dim1 | tile_dim2=0
  g1[5] = (int)stride_elems;                     // tensor_dim0_stride[31:0]
  g1[6] = 0;                                     // stride[47:32] | dim1_stride lo
  g1[7] = 0;
  i32x4 z4 = {0, 0, 0, 0};
  i32x8 z8 = {0, 0, 0, 0, 0, 0, 0, 0};
  // clang-23 / therock-10.0 form: (g0, g1, g2, g3, g4, cpol)
  __builtin_amdgcn_tensor_load_to_lds(g0, g1, z4, z4, z8, 0);
}

// ---------------- f32 -> f16 convert (x) ----------------
__global__ __launch_bounds__(256)
void cvt_f32_f16(const float* __restrict__ in, _Float16* __restrict__ out, int n4) {
  int i = blockIdx.x * blockDim.x + threadIdx.x;
  if (i < n4) {
    f32x4 v = ((const f32x4*)in)[i];
    v4h h;
    h.x = (_Float16)v.x; h.y = (_Float16)v.y;
    h.z = (_Float16)v.z; h.w = (_Float16)v.w;
    *(v4h*)&out[(size_t)i * 4] = h;
  }
}

// ---------------- transpose + convert weights: WT[n][k] = W[k][n] ----------------
__global__ __launch_bounds__(256)
void transpose_cvt(const float* __restrict__ W, _Float16* __restrict__ WT, int N) {
  __shared__ float t[64][65];
  const int bx = blockIdx.x * 64, by = blockIdx.y * 64;
#pragma unroll
  for (int it = 0; it < 16; ++it) {
    int idx = threadIdx.x + it * 256;
    int r = idx >> 6, c = idx & 63;
    t[r][c] = W[(size_t)(by + r) * N + bx + c];
  }
  __syncthreads();
#pragma unroll
  for (int it = 0; it < 16; ++it) {
    int idx = threadIdx.x + it * 256;
    int r = idx >> 6, c = idx & 63;
    WT[(size_t)(bx + r) * N + by + c] = (_Float16)t[c][r];
  }
}

// ---------------- WMMA GEMM: C[M,N] = A[M,K] * B[K,N], B given as BT[N,K] ----------------
// 128x128 tile, 8 waves (2x4 accumulators each), TDM-fed double-buffered LDS.
__global__ __launch_bounds__(256)
void gemm_f16_wmma(const _Float16* __restrict__ A, const _Float16* __restrict__ BT,
                   _Float16* __restrict__ Ch, float* __restrict__ Cf,
                   int M, int N, int K) {
  __shared__ __align__(16) _Float16 As[2][128][40];  // 32-half rows + 16B HW pad
  __shared__ __align__(16) _Float16 Bs[2][128][40];

  const int tid  = threadIdx.x;
  const int wv   = tid >> 5;
  const int lane = tid & 31;
  const int hx   = lane >> 4;
  const int l16  = lane & 15;

  const int row0 = blockIdx.y * 128;
  const int col0 = blockIdx.x * 128;
  const int wr = wv & 3;
  const int wc = wv >> 2;

  v8f acc[2][4];
#pragma unroll
  for (int i = 0; i < 2; ++i)
#pragma unroll
    for (int j = 0; j < 4; ++j) acc[i][j] = v8f_zero();

  auto issue = [&](int buf, int k0) {
    if (wv == 0) {
      // tile = 32 K-elems x 128 rows; row 64B, pad +16B -> 40-half LDS stride
      tdm_load_2d((unsigned)(uintptr_t)&As[buf][0][0], &A[(size_t)row0 * K + k0],
                  32u, 128u, (unsigned)K, (unsigned)M, (unsigned)K, 3u, 3u);
      tdm_load_2d((unsigned)(uintptr_t)&Bs[buf][0][0], &BT[(size_t)col0 * K + k0],
                  32u, 128u, (unsigned)K, (unsigned)N, (unsigned)K, 3u, 3u);
    }
  };

  issue(0, 0);
  for (int k0 = 0; k0 < K; k0 += 32) {
    const int buf = (k0 >> 5) & 1;
    __builtin_amdgcn_s_wait_tensorcnt(0);  // wave0: DMA for `buf` complete
    __syncthreads();                       // publish LDS to all waves
    if (k0 + 32 < K) issue(buf ^ 1, k0 + 32);

    v16h af[2], bf[4];
#pragma unroll
    for (int i = 0; i < 2; ++i) {
      int m = wr * 32 + i * 16 + l16;
      af[i] = ld_frag(&As[buf][m][hx * 8], &As[buf][m][16 + hx * 8]);
    }
#pragma unroll
    for (int j = 0; j < 4; ++j) {
      int n = wc * 64 + j * 16 + l16;
      bf[j] = ld_frag(&Bs[buf][n][hx * 8], &Bs[buf][n][16 + hx * 8]);
    }
#pragma unroll
    for (int i = 0; i < 2; ++i)
#pragma unroll
      for (int j = 0; j < 4; ++j)
        acc[i][j] = __builtin_amdgcn_wmma_f32_16x16x32_f16(
            false, af[i], false, bf[j], (short)0, acc[i][j], false, false);
  }

#pragma unroll
  for (int i = 0; i < 2; ++i)
#pragma unroll
    for (int j = 0; j < 4; ++j)
#pragma unroll
      for (int r = 0; r < 8; ++r) {
        int m = row0 + wr * 32 + i * 16 + r + hx * 8;
        int n = col0 + wc * 64 + j * 16 + l16;
        float v = acc[i][j][r];
        if (Cf) Cf[(size_t)m * N + n] = v;
        else    Ch[(size_t)m * N + n] = (_Float16)v;
      }
}

// ---------------- RoPE + L2 norm + sqk scale; one wave per (s, head) ----------------
__global__ __launch_bounds__(256)
void rope_l2norm(const _Float16* __restrict__ X, const float* __restrict__ fc,
                 const float* __restrict__ fs, const float* __restrict__ sqk,
                 _Float16* __restrict__ Y) {
  const int gw   = blockIdx.x * 8 + (threadIdx.x >> 5);  // s*16 + h
  const int lane = threadIdx.x & 31;
  const int s = gw >> 4, h = gw & 15;
  const int d0 = lane * 4;
  const size_t base = (size_t)s * DIM + h * HEADDIM;

  v4h xv = *(const v4h*)&X[base + d0];
  float v0 = (float)xv.x, v1 = (float)xv.y, v2 = (float)xv.z, v3 = (float)xv.w;
  const int fi = s * (HEADDIM / 2) + (d0 >> 1);
  float c0 = fc[fi],     s0 = fs[fi];
  float c1 = fc[fi + 1], s1 = fs[fi + 1];
  float r0 = v0 * c0 - v1 * s0, i0 = v0 * s0 + v1 * c0;
  float r1 = v2 * c1 - v3 * s1, i1 = v2 * s1 + v3 * c1;

  float ss = r0 * r0 + i0 * i0 + r1 * r1 + i1 * i1;
#pragma unroll
  for (int m = 1; m < 32; m <<= 1) ss += __shfl_xor(ss, m, 32);
  const float inv = 1.f / fmaxf(sqrtf(ss), 1e-12f);
  const float SQ = 45.25483399593904f;  // sqrt(2048) = SQK_INIT_VALUE/SQK_INIT_SCALING
  const float* sq = &sqk[h * HEADDIM + d0];
  v4h o;
  o.x = (_Float16)(r0 * inv * sq[0] * SQ);
  o.y = (_Float16)(i0 * inv * sq[1] * SQ);
  o.z = (_Float16)(r1 * inv * sq[2] * SQ);
  o.w = (_Float16)(i1 * inv * sq[3] * SQ);
  *(v4h*)&Y[base + d0] = o;
}

// ---------------- Flash attention: block = (128 q-rows, 1 head), 8 waves x 16 rows ----
// K tiles DMA'd by the TDM (double-buffered); V staged transposed by the waves.
__global__ __launch_bounds__(256)
void flash_attn(const _Float16* __restrict__ Q, const _Float16* __restrict__ Kt,
                const _Float16* __restrict__ V, _Float16* __restrict__ O, int S) {
  const int h   = blockIdx.y;
  const int q0  = blockIdx.x * 128;
  const int tid = threadIdx.x;
  const int wv = tid >> 5, lane = tid & 31, hx = lane >> 4, l16 = lane & 15;

  __shared__ __align__(16) _Float16 Ks[2][32][136];   // [key][hd], 256B rows +16B HW pad
  __shared__ __align__(16) _Float16 Vs[2][128][40];   // transposed: [hd][key]
  __shared__ __align__(16) _Float16 Ps[8][16][32];    // per-wave P scratch

  auto issueK = [&](int buf, int kk) {
    if (wv == 0)
      tdm_load_2d((unsigned)(uintptr_t)&Ks[buf][0][0],
                  &Kt[(size_t)kk * DIM + h * HEADDIM],
                  128u, 32u, (unsigned)DIM, (unsigned)S, (unsigned)DIM, 5u, 3u);
  };
  auto loadV = [&](int buf, int kk) {
#pragma unroll
    for (int it = 0; it < 2; ++it) {
      int idx = tid + it * 256;
      int r = idx >> 4, ck = (idx & 15) * 8;
      union { f32x4 f; _Float16 e[8]; } u;
      u.f = *(const f32x4*)&V[(size_t)(kk + r) * DIM + h * HEADDIM + ck];
#pragma unroll
      for (int e = 0; e < 8; ++e) Vs[buf][ck + e][r] = u.e[e];
    }
  };

  // Q fragments for this wave's 16 rows (hd split into 4 K=32 chunks)
  v16h qf[4];
  {
    const _Float16* qrow = &Q[(size_t)(q0 + wv * 16 + l16) * DIM + h * HEADDIM];
#pragma unroll
    for (int c = 0; c < 4; ++c)
      qf[c] = ld_frag(&qrow[c * 32 + hx * 8], &qrow[c * 32 + 16 + hx * 8]);
  }

  v8f o[8];
#pragma unroll
  for (int t = 0; t < 8; ++t) o[t] = v8f_zero();
  float mrow[8], lrow[8];
#pragma unroll
  for (int r = 0; r < 8; ++r) { mrow[r] = -1e30f; lrow[r] = 0.f; }
  const float scale = 11.313708498984761f;  // sqrt(hd)

  issueK(0, 0);
  loadV(0, 0);
  for (int kk = 0; kk < S; kk += 32) {
    const int buf = (kk >> 5) & 1;
    __builtin_amdgcn_s_wait_tensorcnt(0);
    __syncthreads();  // K DMA + V stores for `buf` visible to all waves
    if (kk + 32 < S) {
      issueK(buf ^ 1, kk + 32);
      __builtin_prefetch(&V[(size_t)(kk + 32 + (tid >> 4)) * DIM + h * HEADDIM], 0, 0);
      loadV(buf ^ 1, kk + 32);
    }

    // S tile: 16 q-rows x 32 keys as two 16x16 WMMA C-fragments
    v8f st[2];
#pragma unroll
    for (int sub = 0; sub < 2; ++sub) {
      v8f sacc = v8f_zero();
      int key = sub * 16 + l16;
#pragma unroll
      for (int c = 0; c < 4; ++c) {
        v16h kb = ld_frag(&Ks[buf][key][c * 32 + hx * 8],
                          &Ks[buf][key][c * 32 + 16 + hx * 8]);
        sacc = __builtin_amdgcn_wmma_f32_16x16x32_f16(false, qf[c], false, kb,
                                                      (short)0, sacc, false, false);
      }
      st[sub] = sacc;
    }
#pragma unroll
    for (int sub = 0; sub < 2; ++sub)
#pragma unroll
      for (int r = 0; r < 8; ++r) st[sub][r] *= scale;

    // online softmax (row stats replicated across each 16-lane half)
    float newm[8], alpha[8];
#pragma unroll
    for (int r = 0; r < 8; ++r) {
      float v = fmaxf(st[0][r], st[1][r]);
#pragma unroll
      for (int m = 1; m < 16; m <<= 1) v = fmaxf(v, __shfl_xor(v, m, 32));
      newm[r] = fmaxf(mrow[r], v);
      alpha[r] = __expf(mrow[r] - newm[r]);
    }
#pragma unroll
    for (int sub = 0; sub < 2; ++sub)
#pragma unroll
      for (int r = 0; r < 8; ++r) st[sub][r] = __expf(st[sub][r] - newm[r]);
#pragma unroll
    for (int r = 0; r < 8; ++r) {
      float v = st[0][r] + st[1][r];
#pragma unroll
      for (int m = 1; m < 16; m <<= 1) v += __shfl_xor(v, m, 32);
      lrow[r] = lrow[r] * alpha[r] + v;
      mrow[r] = newm[r];
    }
#pragma unroll
    for (int t = 0; t < 8; ++t)
#pragma unroll
      for (int r = 0; r < 8; ++r) o[t][r] *= alpha[r];

    // transpose P (C-fragment layout) -> A-fragment layout via per-wave LDS
#pragma unroll
    for (int sub = 0; sub < 2; ++sub)
#pragma unroll
      for (int r = 0; r < 8; ++r)
        Ps[wv][r + hx * 8][sub * 16 + l16] = (_Float16)st[sub][r];
    v16h pf = ld_frag(&Ps[wv][l16][hx * 8], &Ps[wv][l16][16 + hx * 8]);

    // O += P * V  (8 hd-subtiles)
#pragma unroll
    for (int t = 0; t < 8; ++t) {
      int n = t * 16 + l16;
      v16h vb = ld_frag(&Vs[buf][n][hx * 8], &Vs[buf][n][16 + hx * 8]);
      o[t] = __builtin_amdgcn_wmma_f32_16x16x32_f16(false, pf, false, vb,
                                                    (short)0, o[t], false, false);
    }
  }

#pragma unroll
  for (int t = 0; t < 8; ++t)
#pragma unroll
    for (int r = 0; r < 8; ++r) {
      int m = q0 + wv * 16 + r + hx * 8;
      int n = t * 16 + l16;
      O[(size_t)m * DIM + h * HEADDIM + n] = (_Float16)(o[t][r] / lrow[r]);
    }
}

// ---------------- host launcher ----------------
extern "C" void kernel_launch(void* const* d_in, const int* in_sizes, int n_in,
                              void* d_out, int out_size, void* d_ws, size_t ws_size,
                              hipStream_t stream) {
  (void)in_sizes; (void)n_in; (void)out_size; (void)ws_size;
  const float* x   = (const float*)d_in[0];
  const float* fc  = (const float*)d_in[1];
  const float* fs  = (const float*)d_in[2];
  const float* wq  = (const float*)d_in[3];
  const float* wk  = (const float*)d_in[4];
  const float* wv  = (const float*)d_in[5];
  const float* wo  = (const float*)d_in[6];
  const float* sqk = (const float*)d_in[7];
  float* out = (float*)d_out;

  const size_t SD = (size_t)SEQ * DIM;
  const size_t DD = (size_t)DIM * DIM;
  char* w = (char*)d_ws;
  _Float16* Xh  = (_Float16*)(w);           // x in f16
  _Float16* WqT = (_Float16*)(w + SD * 2);  // transposed f16 weights
  _Float16* WkT = WqT + DD;
  _Float16* WvT = WkT + DD;
  _Float16* WoT = WvT + DD;
  _Float16* Qh  = WoT + DD;                 // raw projections (f16)
  _Float16* Kh  = Qh + SD;
  _Float16* Vh  = Kh + SD;
  _Float16* Qn  = Vh + SD;                  // post rope+norm
  _Float16* Kn  = Qn + SD;
  _Float16* AOh = Kn + SD;                  // attention output (f16)

  cvt_f32_f16<<<(int)(SD / 4 + 255) / 256, 256, 0, stream>>>(x, Xh, (int)(SD / 4));

  dim3 tg(DIM / 64, DIM / 64);
  transpose_cvt<<<tg, 256, 0, stream>>>(wq, WqT, DIM);
  transpose_cvt<<<tg, 256, 0, stream>>>(wk, WkT, DIM);
  transpose_cvt<<<tg, 256, 0, stream>>>(wv, WvT, DIM);
  transpose_cvt<<<tg, 256, 0, stream>>>(wo, WoT, DIM);

  dim3 gg(DIM / 128, SEQ / 128);
  gemm_f16_wmma<<<gg, 256, 0, stream>>>(Xh, WqT, Qh, nullptr, SEQ, DIM, DIM);
  gemm_f16_wmma<<<gg, 256, 0, stream>>>(Xh, WkT, Kh, nullptr, SEQ, DIM, DIM);
  gemm_f16_wmma<<<gg, 256, 0, stream>>>(Xh, WvT, Vh, nullptr, SEQ, DIM, DIM);

  const int nwaves = SEQ * NHEADS;
  rope_l2norm<<<nwaves / 8, 256, 0, stream>>>(Qh, fc, fs, sqk, Qn);
  rope_l2norm<<<nwaves / 8, 256, 0, stream>>>(Kh, fc, fs, sqk, Kn);

  dim3 fg(SEQ / 128, NHEADS);
  flash_attn<<<fg, 256, 0, stream>>>(Qn, Kn, Vh, AOh, SEQ);

  gemm_f16_wmma<<<gg, 256, 0, stream>>>(AOh, WoT, nullptr, out, SEQ, DIM, DIM);
}
